// WritePhaseModel_76708115906949
// MI455X (gfx1250) — matrix operand order repaired and checked by reading, not compiled, not run on hardware
//
#include <hip/hip_runtime.h>
#include <hip/hip_bf16.h>

// ---------------- model constants ----------------
constexpr int   cH   = 2048;
constexpr int   cNH  = 16;
constexpr int   cNKV = 8;
constexpr int   cHD  = 128;
constexpr int   cF   = 5504;
constexpr int   cL   = 2;
constexpr int   cV   = 32000;
constexpr int   cB   = 2;
constexpr int   cT   = 1024;
constexpr int   cM   = 4;
constexpr int   cZD  = 256;
constexpr int   cS   = cM + cT - 1;   // 1027
constexpr int   cBS  = cB * cS;       // 2054
constexpr float cTHETA = 1000000.0f;

// ---------------- vector / WMMA types (POD ext-vectors only) ----------------
typedef __attribute__((ext_vector_type(16))) __bf16        v16bf;
typedef __attribute__((ext_vector_type(8)))  float         v8f;
typedef __attribute__((ext_vector_type(4)))  unsigned int  v4u;
typedef __attribute__((ext_vector_type(8)))  int           v8i;
typedef __attribute__((ext_vector_type(4)))  int           v4i;
typedef __attribute__((ext_vector_type(2)))  float         v2f;

union FragBF {
  v16bf v;
  v4u   q[2];
};

__device__ inline unsigned short f2bf(float x) {
  union { float f; unsigned int u; } c; c.f = x;
  unsigned int u = c.u;
  unsigned int r = u + 0x7FFFu + ((u >> 16) & 1u);          // round-to-nearest-even
  if ((u & 0x7F800000u) == 0x7F800000u) r = u;              // inf/nan passthrough
  return (unsigned short)(r >> 16);
}
__device__ inline unsigned int pack_bf2(float lo, float hi) {
  return (unsigned int)f2bf(lo) | ((unsigned int)f2bf(hi) << 16);
}

// =====================================================================
// Tensor Data Mover: 2D f32 tile (tile_d0 cols x tile_d1 rows) from global
// (row stride = stride0 elements) into LDS at byte offset lds_off.
// tensor_d0/tensor_d1 bound the tensor for OOB handling: reads past the
// tensor return zero -> free zero padding at the M edge.
// D# packed per CDNA5 ISA 8.3-8.6. Issued by one wave; EXEC ignored by HW.
// =====================================================================
__device__ inline void tdm_load_2d_f32(unsigned lds_off, const void* gptr,
                                       int tile_d0, int tile_d1,
                                       int tens_d0, int tens_d1, int stride0) {
  unsigned long long ga = (unsigned long long)(uintptr_t)gptr;
  v4u g0;
  g0[0] = 1u;                                              // count=1, user mode
  g0[1] = lds_off;                                         // lds_addr
  g0[2] = (unsigned int)(ga & 0xFFFFFFFFull);              // global_addr[31:0]
  g0[3] = (unsigned int)((ga >> 32) & 0x01FFFFFFull)       // global_addr[56:32]
        | (2u << 30);                                      // type=2 ("image")
  v8i g1;
  g1[0] = (int)(2u << 16);                                 // data_size=2 (4 bytes)
  g1[1] = (int)(((unsigned)tens_d0 & 0xFFFFu) << 16);      // tensor_dim0[15:0]
  g1[2] = (int)((((unsigned)tens_d0 >> 16) & 0xFFFFu)      // tensor_dim0[31:16]
        | (((unsigned)tens_d1 & 0xFFFFu) << 16));          // tensor_dim1[15:0]
  g1[3] = (int)((((unsigned)tens_d1 >> 16) & 0xFFFFu)      // tensor_dim1[31:16]
        | (((unsigned)tile_d0 & 0xFFFFu) << 16));          // tile_dim0
  g1[4] = (int)((unsigned)tile_d1 & 0xFFFFu);              // tile_dim1, tile_dim2=0
  g1[5] = stride0;                                         // tensor_dim0_stride[31:0]
  g1[6] = 0;
  g1[7] = 0;
  v4i z4 = {0, 0, 0, 0};
#if __has_include(<hip/amd_detail/amd_gfx1250_TDM.h>)
  v8i z8 = {0, 0, 0, 0, 0, 0, 0, 0};
  __builtin_amdgcn_tensor_load_to_lds(g0, g1, z4, z4, z8, 0);
#else
  __builtin_amdgcn_tensor_load_to_lds(g0, g1, z4, z4, 0);
#endif
}

// =====================================================================
// Tiled bf16 WMMA GEMM:  D[M,N] = A[M,K] @ B[K,N]  (+ Cadd if non-null)
// Block = 256 threads = 8 waves; block tile 128x64, K step 32.
// Wave w owns a 16x64 output strip -> 4x v_wmma_f32_16x16x32_bf16 per step.
// Double-buffered TDM pipeline: wave 0 DMAs the NEXT A/B f32 tiles into
// LDS staging while all threads convert the CURRENT tiles to bf16 fragment
// order (so each lane reads its fragment with 2x ds_load_b128) and run WMMA.
// TDM ops complete in-order per wave => s_wait_tensorcnt(2) waits for the
// previous tile pair while the next pair stays in flight.
// Requires N % 64 == 0, K % 32 == 0 (true for all GEMMs here); M edge is
// zero-padded by TDM OOB semantics (tensor_dim1 = M - row0).
// LDS layout (dynamic => known byte offsets for D#.lds_addr):
//   [0,     8192)  A frags : 8 strips * 32 lanes * 16 bf16
//   [8192, 12288)  B frags : 4 subtiles * 32 lanes * 16 bf16
//   [12288,28672)  A stage buf0 : 128x32 f32
//   [28672,45056)  A stage buf1
//   [45056,53248)  B stage buf0 : 32x64 f32
//   [53248,61440)  B stage buf1
// =====================================================================
#define GBM 128
#define GBN 64
#define GBK 32
#define LDS_AFRAG 0u
#define LDS_BFRAG 8192u
#define LDS_AST0  12288u
#define LDS_AST1  28672u
#define LDS_BST0  45056u
#define LDS_BST1  53248u
#define GEMM_LDS_BYTES 61440

__global__ __launch_bounds__(256) void gemm_wmma_bf16_kernel(
    const float* __restrict__ A, const float* __restrict__ Bm,
    const float* __restrict__ Cadd, float* __restrict__ D,
    int M, int N, int K)
{
  extern __shared__ __align__(16) char smem_g[];
  unsigned int* Af32 = (unsigned int*)(smem_g + LDS_AFRAG);  // packed bf16 pairs
  unsigned int* Bf32 = (unsigned int*)(smem_g + LDS_BFRAG);
  const v4u*    Af4  = (const v4u*)(smem_g + LDS_AFRAG);
  const v4u*    Bf4  = (const v4u*)(smem_g + LDS_BFRAG);

  const int tid  = threadIdx.x;
  const int lane = tid & 31;
  const int wave = tid >> 5;            // 0..7 -> M strip
  const int row0 = blockIdx.y * GBM;
  const int col0 = blockIdx.x * GBN;
  const int mrem = M - row0;            // >= 1; TDM zero-pads rows >= mrem

  v8f acc[4] = {v8f{}, v8f{}, v8f{}, v8f{}};

  const unsigned astOff[2] = {LDS_AST0, LDS_AST1};
  const unsigned bstOff[2] = {LDS_BST0, LDS_BST1};
  const int nk = K / GBK;

  // ---- pipeline prologue: DMA tile 0 ----
  if (wave == 0) {
    tdm_load_2d_f32(bstOff[0], Bm + (size_t)0 * N + col0, GBN, GBK, GBN, GBK, N);
    tdm_load_2d_f32(astOff[0], A + (size_t)row0 * K + 0, GBK, GBM, GBK, mrem, K);
  }

  for (int kt = 0; kt < nk; ++kt) {
    const int cur = kt & 1;
    const int nxt = cur ^ 1;
    if (wave == 0) {
      if (kt + 1 < nk) {
        const int k1 = (kt + 1) * GBK;
        tdm_load_2d_f32(bstOff[nxt], Bm + (size_t)k1 * N + col0, GBN, GBK, GBN, GBK, N);
        tdm_load_2d_f32(astOff[nxt], A + (size_t)row0 * K + k1, GBK, GBM, GBK, mrem, K);
        __builtin_amdgcn_s_wait_tensorcnt(2);   // wait for CURRENT pair only
      } else {
        __builtin_amdgcn_s_wait_tensorcnt(0);
      }
    }
    __syncthreads();

    const float* Ast = (const float*)(smem_g + astOff[cur]);
    const float* Bst = (const float*)(smem_g + bstOff[cur]);

    // ---- convert A stage (128x32 f32) -> bf16 fragment order ----
    // ISA 16-bit A layout: element e <-> k = 8*laneHalf + (e<8 ? e : e+8),
    // laneHalf = (k>>3)&1, lane = laneHalf*16 + (row&15).
    for (int i = tid; i < GBM * (GBK / 2); i += 256) {      // 2048 bf16 pairs
      int r = i >> 4;
      int c = (i & 15) << 1;                                // even column
      v2f a2 = *(const v2f*)&Ast[r * GBK + c];              // ds_load_b64
      int strip    = r >> 4;                                // 0..7
      int laneHalf = (c >> 3) & 1;
      int lane_s   = (laneHalf << 4) | (r & 15);
      int e        = ((c >> 4) << 3) | (c & 7);             // even
      Af32[(((strip << 5) | lane_s) << 3) | (e >> 1)] = pack_bf2(a2.x, a2.y);
    }
    // ---- convert B stage (32x64 f32) -> bf16 fragment order ----
    // ISA 16-bit B layout: element e = ((k>>4)<<3)|(k&7), lane = (((k>>3)&1)<<4)|(n&15),
    // subtile = n>>4. Pack pairs along k (even k).
    for (int i = tid; i < (GBK / 2) * GBN; i += 256) {      // 1024 pairs
      int kk = (i >> 6) << 1;                               // even k
      int n  = i & 63;
      float f0 = Bst[kk * GBN + n];
      float f1 = Bst[(kk + 1) * GBN + n];
      int sub    = n >> 4;
      int lane_s = (((kk >> 3) & 1) << 4) | (n & 15);
      int e      = ((kk >> 4) << 3) | (kk & 7);             // even
      Bf32[(((sub << 5) | lane_s) << 3) | (e >> 1)] = pack_bf2(f0, f1);
    }
    __syncthreads();

    // ---- fragment loads: contiguous 32B per lane (2x ds_load_b128 each) ----
    FragBF fa, fb[4];
    {
      int ai = ((wave << 5) | lane) << 1;
      fa.q[0] = Af4[ai];
      fa.q[1] = Af4[ai + 1];
#pragma unroll
      for (int s = 0; s < 4; ++s) {
        int bi = (((s << 5) | lane)) << 1;
        fb[s].q[0] = Bf4[bi];
        fb[s].q[1] = Bf4[bi + 1];
      }
    }
#pragma unroll
    for (int s = 0; s < 4; ++s) {
      acc[s] = __builtin_amdgcn_wmma_f32_16x16x32_bf16(
          false, fa.v, false, fb[s].v, (short)0, acc[s], false, false);
    }
    __syncthreads();
  }

  // ---- store D (ISA f32 C/D layout: VGPR r holds M=r (lanes0-15) / r+8) ----
  const int mbase = row0 + wave * 16 + ((lane >> 4) << 3);
  const int ncb   = col0 + (lane & 15);
#pragma unroll
  for (int s = 0; s < 4; ++s) {
#pragma unroll
    for (int r = 0; r < 8; ++r) {
      int gr = mbase + r;
      if (gr < M) {
        long idx = (long)gr * N + (ncb + s * 16);
        float v0 = acc[s][r];
        if (Cadd) v0 += Cadd[idx];
        D[idx] = v0;
      }
    }
  }
}

// =====================================================================
// z-branch MLP (tiny: M=4)
// =====================================================================
__global__ void zmlp1_kernel(const float* __restrict__ z, const float* __restrict__ w1,
                             const float* __restrict__ b1, float* __restrict__ t1) {
  int i = blockIdx.x * blockDim.x + threadIdx.x;
  if (i >= cM * 2 * cZD) return;
  int r = i / (2 * cZD), c = i % (2 * cZD);
  float s = b1[c];
  for (int k = 0; k < cZD; ++k) s += z[r * cZD + k] * w1[k * (2 * cZD) + c];
  t1[i] = 0.5f * s * (1.0f + erff(s * 0.70710678118654752f));  // exact GELU
}

__global__ void zmlp2_kernel(const float* __restrict__ t1, const float* __restrict__ w2,
                             const float* __restrict__ b2, const float* __restrict__ alpha,
                             float* __restrict__ ze) {
  int i = blockIdx.x * blockDim.x + threadIdx.x;
  if (i >= cM * cH) return;
  int r = i / cH, c = i % cH;
  float s = b2[c];
  for (int k = 0; k < 2 * cZD; ++k) s += t1[r * 2 * cZD + k] * w2[k * cH + c];
  ze[i] = alpha[0] * s;
}

// =====================================================================
// Build x: x[b, 0:M] = z_e ; x[b, M:S] = embed[doc_ids[b, 0:T-1]]
// =====================================================================
__global__ void build_x_kernel(const float* __restrict__ ze, const float* __restrict__ embed,
                               const int* __restrict__ ids, float* __restrict__ x) {
  long i = (long)blockIdx.x * blockDim.x + threadIdx.x;
  if (i >= (long)cBS * cH) return;
  int col = (int)(i % cH);
  long rs = i / cH;
  int s = (int)(rs % cS);
  int b = (int)(rs / cS);
  float v;
  if (s < cM) v = ze[s * cH + col];
  else        v = embed[(long)ids[b * cT + (s - cM)] * cH + col];
  x[i] = v;
}

// =====================================================================
// RMSNorm: one block per row of H=2048
// =====================================================================
__global__ __launch_bounds__(256) void rms_kernel(const float* __restrict__ x,
                                                  const float* __restrict__ w,
                                                  float* __restrict__ out) {
  int row = blockIdx.x;
  const float* xr = x + (long)row * cH;
  __shared__ float red[256];
  float s = 0.0f;
  for (int i = threadIdx.x; i < cH; i += 256) { float v = xr[i]; s += v * v; }
  red[threadIdx.x] = s;
  __syncthreads();
  for (int st = 128; st > 0; st >>= 1) {
    if (threadIdx.x < st) red[threadIdx.x] += red[threadIdx.x + st];
    __syncthreads();
  }
  float r = rsqrtf(red[0] / (float)cH + 1e-6f);
  for (int i = threadIdx.x; i < cH; i += 256)
    out[(long)row * cH + i] = xr[i] * r * w[i];
}

// =====================================================================
// RoPE in-place on [B,S,nheads,HD]; grid = (B*S, nheads), block = 64
// =====================================================================
__global__ void rope_kernel(float* __restrict__ x, int nheads) {
  int bs = blockIdx.x;
  int s  = bs % cS;
  int h  = blockIdx.y;
  int i  = threadIdx.x;  // 0..63
  float inv = __expf(-(2.0f * i / (float)cHD) * __logf(cTHETA));
  float ang = (float)s * inv;
  float c = cosf(ang), sn = sinf(ang);
  long base = ((long)bs * nheads + h) * cHD;
  float x0 = x[base + i];
  float x1 = x[base + i + cHD / 2];
  x[base + i]           = x0 * c - x1 * sn;
  x[base + i + cHD / 2] = x1 * c + x0 * sn;
}

// =====================================================================
// Attention: online-softmax, q-tile = 16, key chunk = 32, GQA rep=2.
// grid = (B*NH, ceil(S/16)), block = 128 (thread t owns output dim t).
// =====================================================================
#define AQT 16
#define AKT 32

__global__ __launch_bounds__(128) void attn_kernel(
    const float* __restrict__ q, const float* __restrict__ k, const float* __restrict__ v,
    const int* __restrict__ amask, float* __restrict__ o)
{
  const int bh = blockIdx.x;
  const int b  = bh / cNH;
  const int h  = bh % cNH;
  const int kvh = h / (cNH / cNKV);
  const int q0 = blockIdx.y * AQT;
  const int t  = threadIdx.x;  // output dim

  __shared__ float qs[AQT][cHD];
  __shared__ float ks[AKT][cHD];
  __shared__ float vs[AKT][cHD];
  __shared__ float sc[AQT][AKT];
  __shared__ float mS[AQT], lS[AQT], scaleS[AQT];

  for (int qi = 0; qi < AQT; ++qi) {
    int qp = q0 + qi;
    qs[qi][t] = (qp < cS) ? q[(((long)b * cS + qp) * cNH + h) * cHD + t] : 0.0f;
  }
  if (t < AQT) { mS[t] = -1e30f; lS[t] = 0.0f; }
  float acc[AQT];
#pragma unroll
  for (int i = 0; i < AQT; ++i) acc[i] = 0.0f;
  __syncthreads();

  int qlast = q0 + AQT - 1; if (qlast > cS - 1) qlast = cS - 1;
  const float rscale = 0.08838834764831845f;  // 1/sqrt(128)

  for (int j0 = 0; j0 <= qlast; j0 += AKT) {
    int jn = qlast - j0 + 1; if (jn > AKT) jn = AKT;
    // stage K,V chunk
    for (int kk = 0; kk < jn; ++kk) {
      long base = (((long)b * cS + (j0 + kk)) * cNKV + kvh) * cHD + t;
      ks[kk][t] = k[base];
      vs[kk][t] = v[base];
    }
    __syncthreads();
    // scores (masked)
    for (int idx = t; idx < AQT * AKT; idx += 128) {
      int qi = idx / AKT, kk = idx % AKT;
      float s;
      if (kk < jn) {
        int qp = q0 + qi, jp = j0 + kk;
        float d = 0.0f;
        for (int c = 0; c < cHD; ++c) d += qs[qi][c] * ks[kk][c];
        s = d * rscale;
        bool km = (jp < cM) ? true : (amask[b * cT + (jp - cM)] != 0);
        bool ok = (jp <= qp) && km && (qp < cS);
        if (!ok) s += -1.0e9f;
      } else {
        s = -1e30f;  // dead slot -> exp ~ 0
      }
      sc[qi][kk] = s;
    }
    __syncthreads();
    // per-query running max / sumexp update (threads 0..15)
    if (t < AQT) {
      float m_old = mS[t];
      float cm = m_old;
      for (int kk = 0; kk < AKT; ++kk) cm = fmaxf(cm, sc[t][kk]);
      float scal = __expf(m_old - cm);
      float ls = lS[t] * scal;
      for (int kk = 0; kk < AKT; ++kk) {
        float w = __expf(sc[t][kk] - cm);
        sc[t][kk] = w;
        ls += w;
      }
      mS[t] = cm; lS[t] = ls; scaleS[t] = scal;
    }
    __syncthreads();
    // accumulate V
#pragma unroll
    for (int qi = 0; qi < AQT; ++qi) {
      float a = acc[qi] * scaleS[qi];
      for (int kk = 0; kk < jn; ++kk) a += sc[qi][kk] * vs[kk][t];
      acc[qi] = a;
    }
    __syncthreads();
  }

  for (int qi = 0; qi < AQT; ++qi) {
    int qp = q0 + qi;
    if (qp < cS)
      o[(((long)b * cS + qp) * cNH + h) * cHD + t] = acc[qi] / fmaxf(lS[qi], 1e-30f);
  }
}

// =====================================================================
// SwiGLU: g = silu(g) * u  (in place)
// =====================================================================
__global__ void swiglu_kernel(float* __restrict__ g, const float* __restrict__ u, long n) {
  long i = (long)blockIdx.x * blockDim.x + threadIdx.x;
  if (i >= n) return;
  float x = g[i];
  g[i] = (x / (1.0f + __expf(-x))) * u[i];
}

// =====================================================================
// Loss: per-row log-softmax NLL over V=32000
// =====================================================================
__global__ __launch_bounds__(256) void loss_rows_kernel(
    const float* __restrict__ logits, const int* __restrict__ ids,
    float* __restrict__ nll, float* __restrict__ vld)
{
  int r = blockIdx.x;                 // 0 .. B*(T-1)-1
  int b = r / (cT - 1), tt = r % (cT - 1);
  long row = (long)b * cS + (cM - 1 + tt);
  const float* lr = logits + row * (long)cV;
  __shared__ float red[256];

  float mx = -1e30f;
  for (int i = threadIdx.x; i < cV; i += 256) mx = fmaxf(mx, lr[i]);
  red[threadIdx.x] = mx;
  __syncthreads();
  for (int st = 128; st > 0; st >>= 1) {
    if (threadIdx.x < st) red[threadIdx.x] = fmaxf(red[threadIdx.x], red[threadIdx.x + st]);
    __syncthreads();
  }
  float gmax = red[0];
  __syncthreads();

  float se = 0.0f;
  for (int i = threadIdx.x; i < cV; i += 256) se += __expf(lr[i] - gmax);
  red[threadIdx.x] = se;
  __syncthreads();
  for (int st = 128; st > 0; st >>= 1) {
    if (threadIdx.x < st) red[threadIdx.x] += red[threadIdx.x + st];
    __syncthreads();
  }

  if (threadIdx.x == 0) {
    int lab = ids[b * cT + tt + 1];
    float lp = lr[lab] - gmax - __logf(red[0]);
    float valid = (lab != 0) ? 1.0f : 0.0f;   // PAD == 0
    nll[r] = -lp * valid;
    vld[r] = valid;
  }
}

__global__ __launch_bounds__(256) void finalize_loss_kernel(
    const float* __restrict__ nll, const float* __restrict__ vld, float* __restrict__ out)
{
  __shared__ float s1[256], s2[256];
  float a = 0.0f, bb = 0.0f;
  for (int i = threadIdx.x; i < cB * (cT - 1); i += 256) { a += nll[i]; bb += vld[i]; }
  s1[threadIdx.x] = a; s2[threadIdx.x] = bb;
  __syncthreads();
  for (int st = 128; st > 0; st >>= 1) {
    if (threadIdx.x < st) { s1[threadIdx.x] += s1[threadIdx.x + st]; s2[threadIdx.x] += s2[threadIdx.x + st]; }
    __syncthreads();
  }
  if (threadIdx.x == 0) out[0] = s1[0] / fmaxf(s2[0], 1.0f);
}

// =====================================================================
// Host orchestration
// =====================================================================
extern "C" void kernel_launch(void* const* d_in, const int* in_sizes, int n_in,
                              void* d_out, int out_size, void* d_ws, size_t ws_size,
                              hipStream_t stream) {
  (void)in_sizes; (void)n_in; (void)out_size; (void)ws_size;

  const float* z_i      = (const float*)d_in[0];
  const int*   doc_ids  = (const int*)  d_in[1];
  const int*   amask    = (const int*)  d_in[2];
  const float* alpha    = (const float*)d_in[3];
  const float* w1       = (const float*)d_in[4];
  const float* b1       = (const float*)d_in[5];
  const float* w2       = (const float*)d_in[6];
  const float* b2       = (const float*)d_in[7];
  const float* embed    = (const float*)d_in[8];
  const float* ln1      = (const float*)d_in[9];
  const float* wq       = (const float*)d_in[10];
  const float* wk       = (const float*)d_in[11];
  const float* wv       = (const float*)d_in[12];
  const float* wo       = (const float*)d_in[13];
  const float* ln2      = (const float*)d_in[14];
  const float* wg       = (const float*)d_in[15];
  const float* wu       = (const float*)d_in[16];
  const float* wd       = (const float*)d_in[17];
  const float* final_ln = (const float*)d_in[18];
  const float* lm_head  = (const float*)d_in[19];

  float* out    = (float*)d_out;
  float* logits = out + 1;            // (loss, logits) flat

  // workspace carve-out
  float* ws = (float*)d_ws;
  size_t off = 0;
  auto carve = [&](size_t n) { float* p = ws + off; off += n; return p; };
  float* x    = carve((size_t)cBS * cH);
  float* nbuf = carve((size_t)cBS * cH);
  float* qb   = carve((size_t)cBS * cNH  * cHD);
  float* kb   = carve((size_t)cBS * cNKV * cHD);
  float* vb   = carve((size_t)cBS * cNKV * cHD);
  float* ob   = carve((size_t)cBS * cNH  * cHD);
  float* gb   = carve((size_t)cBS * cF);
  float* ub   = carve((size_t)cBS * cF);
  float* t1   = carve((size_t)cM * 2 * cZD);
  float* ze   = carve((size_t)cM * cH);
  float* nll  = carve((size_t)cB * (cT - 1));
  float* vld  = carve((size_t)cB * (cT - 1));

  auto gemm = [&](const float* A, const float* Bm, const float* Cadd, float* D,
                  int Mr, int Nc, int Kc) {
    dim3 grid(Nc / GBN, (Mr + GBM - 1) / GBM);
    gemm_wmma_bf16_kernel<<<grid, 256, GEMM_LDS_BYTES, stream>>>(A, Bm, Cadd, D, Mr, Nc, Kc);
  };

  // ---- z branch ----
  zmlp1_kernel<<<(cM * 2 * cZD + 255) / 256, 256, 0, stream>>>(z_i, w1, b1, t1);
  zmlp2_kernel<<<(cM * cH + 255) / 256, 256, 0, stream>>>(t1, w2, b2, alpha, ze);

  // ---- build input sequence ----
  {
    long n = (long)cBS * cH;
    build_x_kernel<<<(unsigned)((n + 255) / 256), 256, 0, stream>>>(ze, embed, doc_ids, x);
  }

  // ---- transformer layers ----
  for (int l = 0; l < cL; ++l) {
    const float* l1 = ln1 + (size_t)l * cH;
    const float* l2 = ln2 + (size_t)l * cH;
    const float* Wq = wq + (size_t)l * cH * (cNH * cHD);
    const float* Wk = wk + (size_t)l * cH * (cNKV * cHD);
    const float* Wv = wv + (size_t)l * cH * (cNKV * cHD);
    const float* Wo = wo + (size_t)l * (cNH * cHD) * cH;
    const float* Wg = wg + (size_t)l * cH * cF;
    const float* Wu = wu + (size_t)l * cH * cF;
    const float* Wd = wd + (size_t)l * cF * cH;

    rms_kernel<<<cBS, 256, 0, stream>>>(x, l1, nbuf);
    gemm(nbuf, Wq, nullptr, qb, cBS, cNH  * cHD, cH);
    gemm(nbuf, Wk, nullptr, kb, cBS, cNKV * cHD, cH);
    gemm(nbuf, Wv, nullptr, vb, cBS, cNKV * cHD, cH);

    rope_kernel<<<dim3(cBS, cNH),  64, 0, stream>>>(qb, cNH);
    rope_kernel<<<dim3(cBS, cNKV), 64, 0, stream>>>(kb, cNKV);

    attn_kernel<<<dim3(cB * cNH, (cS + AQT - 1) / AQT), 128, 0, stream>>>(qb, kb, vb, amask, ob);

    gemm(ob, Wo, x, x, cBS, cH, cNH * cHD);   // h += o @ Wo

    rms_kernel<<<cBS, 256, 0, stream>>>(x, l2, nbuf);
    gemm(nbuf, Wg, nullptr, gb, cBS, cF, cH);
    gemm(nbuf, Wu, nullptr, ub, cBS, cF, cH);
    {
      long n = (long)cBS * cF;
      swiglu_kernel<<<(unsigned)((n + 255) / 256), 256, 0, stream>>>(gb, ub, n);
    }
    gemm(gb, Wd, x, x, cBS, cH, cF);          // h += swiglu @ Wd
  }

  // ---- head ----
  rms_kernel<<<cBS, 256, 0, stream>>>(x, final_ln, nbuf);
  gemm(nbuf, lm_head, nullptr, logits, cBS, cV, cH);

  // ---- loss ----
  loss_rows_kernel<<<cB * (cT - 1), 256, 0, stream>>>(logits, doc_ids, nll, vld);
  finalize_loss_kernel<<<1, 256, 0, stream>>>(nll, vld, out);
}
